// GATConvLayer_new_16389595202404
// MI455X (gfx1250) — matrix operand order repaired and checked by reading, not compiled, not run on hardware
//
#include <hip/hip_runtime.h>
#include <math.h>

// ---------------- CDNA5 WMMA types / helpers ----------------
typedef __attribute__((ext_vector_type(16))) __bf16       bf16x16;
typedef __attribute__((ext_vector_type(8)))  float        f32x8;
typedef __attribute__((ext_vector_type(4)))  unsigned int u32x4;

struct Frag32B { u32x4 lo, hi; };

__device__ inline bf16x16 mkfrag(u32x4 lo, u32x4 hi) {
    Frag32B t{lo, hi};
    return __builtin_bit_cast(bf16x16, t);
}
__device__ inline bf16x16 ldfrag(const unsigned short* p) {
    const u32x4* q = (const u32x4*)p;  // 32B-aligned packed fragment
    return mkfrag(q[0], q[1]);
}
__device__ inline f32x8 zero8() {
    f32x8 z = {0.f,0.f,0.f,0.f,0.f,0.f,0.f,0.f};
    return z;
}
#define WMMA_BF16(A,B,C) \
    __builtin_amdgcn_wmma_f32_16x16x32_bf16(false,(A),false,(B),(short)0,(C),false,false)

__device__ inline unsigned short f2bf(float f) {
    unsigned int u = __builtin_bit_cast(unsigned int, f);
    unsigned int r = u + 0x7FFFu + ((u >> 16) & 1u);   // round-to-nearest-even
    return (unsigned short)(r >> 16);
}
__device__ inline u32x4 pack8(const float4& p0, const float4& p1) {
    u32x4 r;
    r[0] = (unsigned)f2bf(p0.x) | ((unsigned)f2bf(p0.y) << 16);
    r[1] = (unsigned)f2bf(p0.z) | ((unsigned)f2bf(p0.w) << 16);
    r[2] = (unsigned)f2bf(p1.x) | ((unsigned)f2bf(p1.y) << 16);
    r[3] = (unsigned)f2bf(p1.z) | ((unsigned)f2bf(p1.w) << 16);
    return r;
}
__device__ inline float lrelu(float v) { return v >= 0.f ? v : 0.1f * v; }

#define NN 512
#define NE (NN * NN)            // 262144 edges
#define EPSB 1e-5f

// ---------------- workspace layout (bytes) ----------------
static constexpr size_t OFF_XB  = 0;        // x bf16          512*64*2  = 65536
static constexpr size_t OFF_PW1 = 65536;    // W_h2 frags      25*512*2  = 25600
static constexpr size_t OFF_PW2 = 91136;    // W_eo frags       6*512*2  = 6144
static constexpr size_t OFF_XN  = 97280;    // xn f32          512*64*4  = 131072
static constexpr size_t OFF_PXN = 228352;   // xn B-frags      64*512*2  = 65536
static constexpr size_t OFF_A   = 293888;   // a[i] f32        512*4
static constexpr size_t OFF_C   = 295936;   // c[j] f32        512*4
static constexpr size_t OFF_ST  = 297984;   // col sums f32    32*4  (+32*4 sumsq right after)
static constexpr size_t OFF_ST2 = 298112;
static constexpr size_t OFF_EB  = 298240;   // eb f32          262144*4 = 1048576
static constexpr size_t OFF_NX  = 1346816;  // new_x pre-BN    512*64*4 = 131072
// total ~1.41 MB

// ---------------- kernels ----------------

__global__ void __launch_bounds__(64) k_init_stats(float* stats) {
    stats[threadIdx.x] = 0.f;   // zeros both sum[32] and sumsq[32]
}

// pack W_h2 / W_eo into per-lane B-fragment layout; convert x -> bf16
__global__ void __launch_bounds__(256) k_setup1(
    const float* __restrict__ x, const float* __restrict__ W_h2,
    const float* __restrict__ W_eo, unsigned short* __restrict__ xb,
    unsigned short* __restrict__ pW1, unsigned short* __restrict__ pW2)
{
    int idx = blockIdx.x * 256 + threadIdx.x;
    if (idx < 12800) {                       // 25 frags of W_h2^T [160 x 80]
        int f = idx >> 9, s = idx & 511, lane = s >> 4, e = s & 15;
        int kk = f / 5, nt = f - 5 * kk;
        int K   = 32 * kk + (e & 7) + 16 * (e >> 3) + 8 * (lane >> 4);
        int col = 16 * nt + (lane & 15);
        pW1[idx] = f2bf(W_h2[col * 160 + K]);
    } else if (idx < 15872) {                // 6 frags of W_eo^T [80->96 x 32]
        int q = idx - 12800;
        int f = q >> 9, s = q & 511, lane = s >> 4, e = s & 15;
        int kk = f >> 1, nt = f & 1;
        int K   = 32 * kk + (e & 7) + 16 * (e >> 3) + 8 * (lane >> 4);
        int col = 16 * nt + (lane & 15);
        pW2[q] = (K < 80) ? f2bf(W_eo[col * 80 + K]) : (unsigned short)0;
    } else if (idx < 15872 + 32768) {        // x -> bf16
        int q = idx - 15872;
        xb[q] = f2bf(x[q]);
    }
}

// xn = lrelu(x @ W_nu^T + b_nu)   (tiny: 2M MAC, scalar fp32)
__global__ void __launch_bounds__(256) k_xn(
    const float* __restrict__ x, const float* __restrict__ W_nu,
    const float* __restrict__ b_nu, float* __restrict__ xn)
{
    int idx = blockIdx.x * 256 + threadIdx.x;   // 32768 outputs
    int i = idx >> 6, cc = idx & 63;
    float s = b_nu[cc];
    for (int k = 0; k < 64; k++) s += x[i * 64 + k] * W_nu[cc * 64 + k];
    xn[idx] = lrelu(s);
}

// pack xn as B-fragments [512 K x 64 N]; compute a[i]=xn.ev[0:64], c[j]=xn.ev[96:160]
__global__ void __launch_bounds__(256) k_setup2(
    const float* __restrict__ xn, const float* __restrict__ evec,
    unsigned short* __restrict__ pXn, float* __restrict__ a_vec,
    float* __restrict__ c_vec)
{
    int idx = blockIdx.x * 256 + threadIdx.x;
    if (idx < 32768) {                       // 64 frags (kk 0..15, nt 0..3)
        int f = idx >> 9, s = idx & 511, lane = s >> 4, e = s & 15;
        int kk = f >> 2, nt = f & 3;
        int K   = 32 * kk + (e & 7) + 16 * (e >> 3) + 8 * (lane >> 4);
        int col = 16 * nt + (lane & 15);
        pXn[idx] = f2bf(xn[K * 64 + col]);
    } else if (idx < 32768 + 512) {
        int i = idx - 32768;
        float sa = 0.f, sc = 0.f;
        for (int k = 0; k < 64; k++) {
            float v = xn[i * 64 + k];
            sa += v * evec[k];
            sc += v * evec[96 + k];
        }
        a_vec[i] = sa; c_vec[i] = sc;
    }
}

// Fused edge pipeline: 32 edges (2 row-tiles) per wave; tiles share source node i,
// so src A-frags are shared and every B-fragment feeds 2 WMMAs.
// GEMM1 [2x 16x160]@[160x80] (50 WMMA) -> lrelu -> LDS reshape ->
// GEMM2 [2x 16x96]@[96x32]  (12 WMMA) -> lrelu*mask -> store pre-BN edge output.
__global__ void __launch_bounds__(32) k_edge(
    const unsigned short* __restrict__ xb, const float* __restrict__ ea,
    const int* __restrict__ mask, const unsigned short* __restrict__ pW1,
    const unsigned short* __restrict__ pW2, const float* __restrict__ b_h2,
    const float* __restrict__ b_eo, float* __restrict__ outE)
{
    __shared__ unsigned short hl[2][16 * 96];   // h tiles, bf16, K padded 80->96
    const int lane = threadIdx.x;
    const int half = lane >> 4, m = lane & 15;
    const int rb = blockIdx.x << 5;          // edge row base (32 rows)
    const int i  = rb >> 9;                  // source node (constant across stripe)
    const int jb = rb & (NN - 1);            // dest-node base

    // ---- A fragments (16-bit A layout: K = (e&7) + 16*(e>>3) + 8*(lane>>4)) ----
    bf16x16 asrc[2];                         // shared by both tiles (broadcast rows)
    {
        const unsigned short* xr = xb + i * 64;
        #pragma unroll
        for (int kk = 0; kk < 2; kk++) {
            int fb = 32 * kk + 8 * half;
            asrc[kk] = mkfrag(*(const u32x4*)(xr + fb), *(const u32x4*)(xr + fb + 16));
        }
    }
    bf16x16 aea[2], adst[2][2];              // per-tile edge / dst fragments
    #pragma unroll
    for (int t = 0; t < 2; t++) {
        const float* er = ea + (size_t)(rb + 16 * t + m) * 32 + 8 * half;
        float4 a0 = *(const float4*)(er);
        float4 a1 = *(const float4*)(er + 4);
        float4 a2 = *(const float4*)(er + 16);
        float4 a3 = *(const float4*)(er + 20);
        aea[t] = mkfrag(pack8(a0, a1), pack8(a2, a3));
        const unsigned short* xj = xb + (size_t)(jb + 16 * t + m) * 64;
        #pragma unroll
        for (int kk = 0; kk < 2; kk++) {
            int fb = 32 * kk + 8 * half;
            adst[t][kk] = mkfrag(*(const u32x4*)(xj + fb), *(const u32x4*)(xj + fb + 16));
        }
    }

    // ---- GEMM1: acc[t][n] covers tile t, output cols 16n..16n+15 ----
    f32x8 acc[2][5];
    #pragma unroll
    for (int t = 0; t < 2; t++)
        #pragma unroll
        for (int n = 0; n < 5; n++) acc[t][n] = zero8();
    #pragma unroll
    for (int kk = 0; kk < 5; kk++) {
        #pragma unroll
        for (int n = 0; n < 5; n++) {
            bf16x16 b = ldfrag(pW1 + (size_t)(kk * 5 + n) * 512 + lane * 16);
            #pragma unroll
            for (int t = 0; t < 2; t++) {
                bf16x16 A = (kk < 2) ? asrc[kk] : (kk == 2 ? aea[t] : adst[t][kk - 3]);
                acc[t][n] = WMMA_BF16(A, b, acc[t][n]);
            }
        }
    }

    // ---- bias + lrelu, stage h into LDS (zero-padded K 80..95) ----
    {
        unsigned int* z = (unsigned int*)&hl[0][0];
        #pragma unroll
        for (int q = 0; q < 48; q++) z[lane + 32 * q] = 0u;   // 1536 dwords
    }
    __syncthreads();
    #pragma unroll
    for (int t = 0; t < 2; t++)
        #pragma unroll
        for (int n = 0; n < 5; n++) {
            float bias = b_h2[16 * n + m];
            #pragma unroll
            for (int r = 0; r < 8; r++) {
                float v = lrelu(acc[t][n][r] + bias);
                hl[t][(r + 8 * half) * 96 + 16 * n + m] = f2bf(v);   // [row M][col]
            }
        }
    __syncthreads();

    // ---- GEMM2: h[16x96] @ W_eo^T[96x32], both tiles share each B-fragment ----
    f32x8 acc2[2][2];
    acc2[0][0] = zero8(); acc2[0][1] = zero8();
    acc2[1][0] = zero8(); acc2[1][1] = zero8();
    #pragma unroll
    for (int kk = 0; kk < 3; kk++) {
        bf16x16 a2[2];
        #pragma unroll
        for (int t = 0; t < 2; t++) {
            const unsigned short* hp = &hl[t][0] + m * 96 + 32 * kk + 8 * half;
            a2[t] = mkfrag(*(const u32x4*)hp, *(const u32x4*)(hp + 16));
        }
        #pragma unroll
        for (int nt = 0; nt < 2; nt++) {
            bf16x16 b = ldfrag(pW2 + (size_t)(kk * 2 + nt) * 512 + lane * 16);
            #pragma unroll
            for (int t = 0; t < 2; t++)
                acc2[t][nt] = WMMA_BF16(a2[t], b, acc2[t][nt]);
        }
    }

    // ---- bias + lrelu + mask, store pre-BN edge output ----
    #pragma unroll
    for (int t = 0; t < 2; t++) {
        int rbt = rb + 16 * t;
        u32x4 mq0 = *(const u32x4*)(mask + rbt + 8 * half);
        u32x4 mq1 = *(const u32x4*)(mask + rbt + 8 * half + 4);
        float mval[8];
        #pragma unroll
        for (int r = 0; r < 4; r++) {
            mval[r]     = (float)(int)mq0[r];
            mval[4 + r] = (float)(int)mq1[r];
        }
        #pragma unroll
        for (int nt = 0; nt < 2; nt++) {
            int col = 16 * nt + m;
            float bias = b_eo[col];
            #pragma unroll
            for (int r = 0; r < 8; r++) {
                float v = lrelu(acc2[t][nt][r] + bias) * mval[r];
                outE[(size_t)(rbt + r + 8 * half) * 32 + col] = v;
            }
        }
    }
}

// eb[e] = nea[e].ev[64:96] (wave reduce) + per-column sum/sumsq partials -> atomics
__global__ void __launch_bounds__(256) k_ebstats(
    const float* __restrict__ outE, const float* __restrict__ evec,
    float* __restrict__ eb, float* __restrict__ stats, float* __restrict__ stats2)
{
    __shared__ float sls[8][32], sl2[8][32];
    int lane = threadIdx.x & 31, w = threadIdx.x >> 5;
    int wg = blockIdx.x * 8 + w;
    int nw = gridDim.x * 8;
    float ev = evec[64 + lane];
    float ls = 0.f, l2 = 0.f;
    for (int r = wg; r < NE; r += nw) {
        float v = outE[(size_t)r * 32 + lane];
        ls += v; l2 += v * v;
        float red = v * ev;
        for (int off = 16; off; off >>= 1) red += __shfl_xor(red, off, 32);
        if (lane == 0) eb[r] = red;
    }
    sls[w][lane] = ls; sl2[w][lane] = l2;
    __syncthreads();
    if (threadIdx.x < 32) {
        float a = 0.f, b = 0.f;
        for (int g = 0; g < 8; g++) { a += sls[g][threadIdx.x]; b += sl2[g][threadIdx.x]; }
        atomicAdd(&stats[threadIdx.x], a);
        atomicAdd(&stats2[threadIdx.x], b);
    }
}

// scores -> softmax -> *mask -> att(bf16, LDS) -> att @ xn via WMMA
__global__ void __launch_bounds__(256) k_att(
    const float* __restrict__ a_vec, const float* __restrict__ c_vec,
    const float* __restrict__ eb, const int* __restrict__ mask,
    const unsigned short* __restrict__ pXn, float* __restrict__ newx)
{
    __shared__ unsigned short att[16 * 520];   // 8-elem row pad kills LDS bank conflicts
    const int lane = threadIdx.x & 31, w = threadIdx.x >> 5;
    const int half = lane >> 4, m = lane & 15;
    const int ib = blockIdx.x << 4;

    // phase 1: each of 8 waves softmaxes 2 rows
    #pragma unroll
    for (int rr = 0; rr < 2; rr++) {
        int row = 2 * w + rr;
        int i = ib + row;
        float ai = a_vec[i];
        float s[16];
        float mx = -3.4e38f;
        #pragma unroll
        for (int t = 0; t < 16; t++) {
            int j = lane + 32 * t;
            float v = lrelu(ai + eb[(size_t)i * 512 + j] + c_vec[j]);
            s[t] = v; mx = fmaxf(mx, v);
        }
        for (int off = 16; off; off >>= 1) mx = fmaxf(mx, __shfl_xor(mx, off, 32));
        float sum = 0.f;
        #pragma unroll
        for (int t = 0; t < 16; t++) { s[t] = __expf(s[t] - mx); sum += s[t]; }
        for (int off = 16; off; off >>= 1) sum += __shfl_xor(sum, off, 32);
        float inv = 1.f / sum;
        #pragma unroll
        for (int t = 0; t < 16; t++) {
            int j = lane + 32 * t;
            float av = s[t] * inv * (float)mask[(size_t)i * 512 + j];
            att[row * 520 + j] = f2bf(av);
        }
    }
    __syncthreads();

    // phase 2: waves 0..3 each own 16 output columns; 16 WMMAs over K=512
    if (w < 4) {
        int nt = w;
        f32x8 acc = zero8();
        for (int kk = 0; kk < 16; kk++) {
            const unsigned short* ap = att + m * 520 + 32 * kk + 8 * half;
            bf16x16 af = mkfrag(*(const u32x4*)ap, *(const u32x4*)(ap + 16));
            bf16x16 bf = ldfrag(pXn + (size_t)(kk * 4 + nt) * 512 + lane * 16);
            acc = WMMA_BF16(af, bf, acc);
        }
        #pragma unroll
        for (int r = 0; r < 8; r++)
            newx[(size_t)(ib + r + 8 * half) * 64 + 16 * nt + m] = acc[r];
    }
}

// BatchNorm of new_x (512 x 64): single block, batch stats, biased variance
__global__ void __launch_bounds__(1024) k_bn_x(
    const float* __restrict__ newx, const float* __restrict__ bw,
    const float* __restrict__ bb, float* __restrict__ out)
{
    __shared__ float ss[1024], sq[1024];
    __shared__ float sc[64], sh[64];
    int t = threadIdx.x;
    int col = t & 63, g = t >> 6;            // 16 threads per column
    float a = 0.f, b = 0.f;
    for (int k = 0; k < 32; k++) {
        float v = newx[(size_t)(g + 16 * k) * 64 + col];
        a += v; b += v * v;
    }
    ss[t] = a; sq[t] = b;
    __syncthreads();
    if (t < 64) {
        float s = 0.f, s2 = 0.f;
        for (int gg = 0; gg < 16; gg++) { s += ss[t + 64 * gg]; s2 += sq[t + 64 * gg]; }
        float mean = s * (1.f / 512.f);
        float var  = s2 * (1.f / 512.f) - mean * mean;
        float scale = bw[t] * rsqrtf(var + EPSB);
        sc[t] = scale; sh[t] = bb[t] - mean * scale;
    }
    __syncthreads();
    float scale = sc[col], shift = sh[col];
    for (int k = 0; k < 32; k++) {
        size_t idx = (size_t)(g + 16 * k) * 64 + col;
        out[idx] = newx[idx] * scale + shift;
    }
}

// BatchNorm of new_edge_attr in place using reduced stats
__global__ void __launch_bounds__(256) k_bn_e(
    const float* __restrict__ stats, const float* __restrict__ stats2,
    const float* __restrict__ bw, const float* __restrict__ bb,
    float* __restrict__ outE)
{
    size_t idx = (size_t)blockIdx.x * 256 + threadIdx.x;
    size_t stride = (size_t)gridDim.x * 256;
    const float invM = 1.f / (float)NE;
    for (size_t p = idx; p < (size_t)NE * 32; p += stride) {
        int col = (int)(p & 31);
        float mean = stats[col] * invM;
        float var  = stats2[col] * invM - mean * mean;
        float scale = bw[col] * rsqrtf(var + EPSB);
        float shift = bb[col] - mean * scale;
        outE[p] = outE[p] * scale + shift;
    }
}

// ---------------- launcher ----------------
extern "C" void kernel_launch(void* const* d_in, const int* in_sizes, int n_in,
                              void* d_out, int out_size, void* d_ws, size_t ws_size,
                              hipStream_t stream)
{
    (void)in_sizes; (void)n_in; (void)out_size; (void)ws_size;
    const float* x    = (const float*)d_in[0];
    const float* ea   = (const float*)d_in[1];
    const int*   mask = (const int*)d_in[2];
    const float* W_nu = (const float*)d_in[3];
    const float* b_nu = (const float*)d_in[4];
    const float* W_h2 = (const float*)d_in[5];
    const float* b_h2 = (const float*)d_in[6];
    const float* W_eo = (const float*)d_in[7];
    const float* b_eo = (const float*)d_in[8];
    const float* evec = (const float*)d_in[9];
    const float* bnw  = (const float*)d_in[10];
    const float* bnb  = (const float*)d_in[11];
    const float* bew  = (const float*)d_in[12];
    const float* beb  = (const float*)d_in[13];

    char* ws = (char*)d_ws;
    unsigned short* xb   = (unsigned short*)(ws + OFF_XB);
    unsigned short* pW1  = (unsigned short*)(ws + OFF_PW1);
    unsigned short* pW2  = (unsigned short*)(ws + OFF_PW2);
    float*          xn   = (float*)(ws + OFF_XN);
    unsigned short* pXn  = (unsigned short*)(ws + OFF_PXN);
    float*          avec = (float*)(ws + OFF_A);
    float*          cvec = (float*)(ws + OFF_C);
    float*          st   = (float*)(ws + OFF_ST);
    float*          st2  = (float*)(ws + OFF_ST2);
    float*          eb   = (float*)(ws + OFF_EB);
    float*          nx   = (float*)(ws + OFF_NX);

    float* outX = (float*)d_out;
    float* outE = (float*)d_out + 512 * 64;

    k_init_stats<<<1, 64, 0, stream>>>(st);                       // zeros st + st2
    k_setup1<<<190, 256, 0, stream>>>(x, W_h2, W_eo, xb, pW1, pW2);
    k_xn<<<128, 256, 0, stream>>>(x, W_nu, b_nu, xn);
    k_setup2<<<130, 256, 0, stream>>>(xn, evec, pXn, avec, cvec);
    k_edge<<<NE / 32, 32, 0, stream>>>(xb, ea, mask, pW1, pW2, b_h2, b_eo, outE);
    k_ebstats<<<512, 256, 0, stream>>>(outE, evec, eb, st, st2);
    k_att<<<NN / 16, 256, 0, stream>>>(avec, cvec, eb, mask, pXn, nx);
    k_bn_x<<<1, 1024, 0, stream>>>(nx, bnw, bnb, outX);
    k_bn_e<<<4096, 256, 0, stream>>>(st, st2, bew, beb, outE);
}
